// MultiHeadAttention_83623013253432
// MI455X (gfx1250) — compile-verified, matrix operand order checked by reading, not run on hardware
//
#include <hip/hip_runtime.h>
#include <hip/hip_bf16.h>
#include <math.h>

// ---------------------------------------------------------------------------
// MultiHeadAttention for MI455X (gfx1250, wave32, WMMA + async LDS staging)
//   B=2, N=M=2048, D=1024, H=16, DH=64
// Pipeline:
//   gemm_bias_wmma<f32,headsplit>  x3  -> Q,K,V (f16, [B,H,S,DH])
//   attn_wmma (flash attention, online softmax, async K staging) -> (B,N,D) f16
//   gemm_bias_wmma<f16,flat> (async A staging) -> out = weighted @ Wp + bp
// ---------------------------------------------------------------------------

typedef _Float16 half_t;
typedef __attribute__((ext_vector_type(16))) _Float16 v16h;
typedef __attribute__((ext_vector_type(8)))  float    v8f;

#define BATCH 2
#define SEQ   2048
#define DMODEL 1024
#define NHEAD 16
#define DHEAD 64
#define SCALE 0.125f   // 64^-0.5

// ---- async global -> LDS copy (CDNA5 GLOBAL_LOAD_ASYNC_TO_LDS, ASYNCcnt) ---
__device__ __forceinline__ void async_ld_b128(unsigned lds_off, const void* gaddr) {
    asm volatile("global_load_async_to_lds_b128 %0, %1, off"
                 :: "v"(lds_off), "v"((unsigned long long)(size_t)gaddr)
                 : "memory");
}
__device__ __forceinline__ void wait_async0() {
    asm volatile("s_wait_asynccnt 0" ::: "memory");
}

// ---- WMMA fragment helpers (16-bit A/B layout, ISA 7.12.2) -----------------
// lane 0-15: row M=lane, K = {0..7, 16..23}; lane 16-31: row M=lane-16,
// K = {8..15, 24..31}. Packed pairs -> two 16-byte chunks per fragment.
__device__ __forceinline__ v16h frag_ld(const half_t* p) {
    v16h f;
    ((uint4*)&f)[0] = *(const uint4*)(p);
    ((uint4*)&f)[1] = *(const uint4*)(p + 16);
    return f;
}
__device__ __forceinline__ const half_t* frag_ptr(const half_t* base, int row0,
                                                  int stride, int koff, int lane) {
    return base + (size_t)(row0 + (lane & 15)) * stride + koff + ((lane >> 4) << 3);
}

__device__ __forceinline__ v8f wmma_f16(v16h a, v16h b, v8f c) {
    return __builtin_amdgcn_wmma_f32_16x16x32_f16(false, a, false, b,
                                                  (short)0, c, false, false);
}

// half-wave (16 lane) reductions: xor masks 1..8 stay within each 16-lane half,
// matching the C/D VGPR layout (lanes 0-15 rows r, lanes 16-31 rows r+8).
__device__ __forceinline__ float hmax16(float x) {
    #pragma unroll
    for (int m = 1; m < 16; m <<= 1) x = fmaxf(x, __shfl_xor(x, m, 32));
    return x;
}
__device__ __forceinline__ float hsum16(float x) {
    #pragma unroll
    for (int m = 1; m < 16; m <<= 1) x += __shfl_xor(x, m, 32);
    return x;
}

// ---------------------------------------------------------------------------
// GEMM: out[S x 1024] = X[S x 1024] @ W[1024 x 1024] + bias
// Block tile 128x128, K-step 32, 256 threads (8 waves), wave tile 32x64.
// ---------------------------------------------------------------------------
#define ASTR 48   // 32 + 16 halves pad; 96B rows keep 16B alignment

template <typename InT, typename OutT, bool HEADSPLIT>
__global__ __launch_bounds__(256) void gemm_bias_wmma(
    const InT* __restrict__ X, const float* __restrict__ W,
    const float* __restrict__ bias, OutT* __restrict__ out) {

    __shared__ __attribute__((aligned(16))) half_t lds_a[128 * ASTR];
    __shared__ __attribute__((aligned(16))) half_t lds_b[128 * ASTR];

    const int tid   = threadIdx.x;
    const int lane  = tid & 31;
    const int w     = tid >> 5;
    const int colid = lane & 15;
    const int hs    = lane >> 4;
    const int wave_m0 = (w & 3) * 32;   // 4 wave-rows x 32
    const int wave_n0 = (w >> 2) * 64;  // 2 wave-cols x 64
    const int row_blk = blockIdx.y * 128;
    const int col_blk = blockIdx.x * 128;

    v8f acc[2][4] = {};

    for (int k0 = 0; k0 < DMODEL; k0 += 32) {
        // stage A tile (128x32)
        if constexpr (sizeof(InT) == 2) {
            // X already f16: async DMA straight into LDS (no conversion needed)
            #pragma unroll
            for (int i = 0; i < 2; ++i) {
                int idx = i * 256 + tid;
                int ar = idx >> 2, seg = idx & 3;    // 4 x 16B per 64B row
                async_ld_b128((unsigned)(size_t)&lds_a[ar * ASTR + seg * 8],
                              X + (size_t)(row_blk + ar) * DMODEL + k0 + seg * 8);
            }
        } else {
            #pragma unroll
            for (int i = 0; i < 16; ++i) {
                int idx = i * 256 + tid;
                int ar = idx >> 5, ac = idx & 31;
                lds_a[ar * ASTR + ac] =
                    (half_t)X[(size_t)(row_blk + ar) * DMODEL + k0 + ac];
            }
        }
        // stage B tile (32x128), transposed in LDS so frag reads are contiguous
        #pragma unroll
        for (int i = 0; i < 16; ++i) {
            int idx = i * 256 + tid;
            int bk = idx >> 7, bn = idx & 127;
            lds_b[bn * ASTR + bk] =
                (half_t)W[(size_t)(k0 + bk) * DMODEL + col_blk + bn];
        }
        if (k0 + 32 < DMODEL)  // prefetch next W tile -> global_prefetch_b8
            __builtin_prefetch(&W[(size_t)(k0 + 32) * DMODEL + col_blk + tid], 0, 1);
        if constexpr (sizeof(InT) == 2) wait_async0();
        __syncthreads();

        v16h af[2], bf[4];
        #pragma unroll
        for (int mt = 0; mt < 2; ++mt)
            af[mt] = frag_ld(frag_ptr(lds_a, wave_m0 + mt * 16, ASTR, 0, lane));
        #pragma unroll
        for (int nt = 0; nt < 4; ++nt)
            bf[nt] = frag_ld(frag_ptr(lds_b, wave_n0 + nt * 16, ASTR, 0, lane));
        #pragma unroll
        for (int mt = 0; mt < 2; ++mt)
            #pragma unroll
            for (int nt = 0; nt < 4; ++nt)
                acc[mt][nt] = wmma_f16(af[mt], bf[nt], acc[mt][nt]);
        __syncthreads();
    }

    // epilogue: bias add, store
    #pragma unroll
    for (int nt = 0; nt < 4; ++nt) {
        const int gn = col_blk + wave_n0 + nt * 16 + colid;
        const float bv = bias[gn];
        #pragma unroll
        for (int mt = 0; mt < 2; ++mt) {
            const float* af = (const float*)&acc[mt][nt];
            #pragma unroll
            for (int r = 0; r < 8; ++r) {
                const int gm = row_blk + wave_m0 + mt * 16 + r + hs * 8;
                const float val = af[r] + bv;
                if (HEADSPLIT) {
                    const int b = gm >> 11, s = gm & 2047;
                    const int h = gn >> 6,  dh = gn & 63;
                    ((half_t*)out)[(((size_t)(b * NHEAD + h) * SEQ + s) << 6) + dh] =
                        (half_t)val;
                } else {
                    ((float*)out)[(size_t)gm * DMODEL + gn] = val;
                }
            }
        }
    }
}

// ---------------------------------------------------------------------------
// Flash attention: block = (b, h, 64 query rows), 128 threads / 4 waves,
// each wave owns 16 query rows. KV tiles of 64, online softmax.
// K tile staged via async DMA (ASYNCcnt); V staged transposed by VALU.
// ---------------------------------------------------------------------------
#define KSTR 72   // 64 + 8 halves pad; 144B rows keep 16B alignment

__global__ __launch_bounds__(128) void attn_wmma(
    const half_t* __restrict__ Q, const half_t* __restrict__ K,
    const half_t* __restrict__ V, const unsigned char* __restrict__ mask,
    half_t* __restrict__ Wout) {

    __shared__ __attribute__((aligned(16))) half_t lds_k[64 * KSTR];
    __shared__ __attribute__((aligned(16))) half_t lds_vt[64 * KSTR];
    __shared__ __attribute__((aligned(16))) half_t lds_p[4 * 16 * KSTR];

    const int tid   = threadIdx.x;
    const int lane  = tid & 31;
    const int w     = tid >> 5;
    const int colid = lane & 15;
    const int hs    = lane >> 4;
    const int nblk  = blockIdx.x, hh = blockIdx.y, bb = blockIdx.z;

    const size_t head_base = (size_t)(bb * NHEAD + hh) * SEQ * DHEAD;
    const half_t* qp = Q + head_base;
    const half_t* kp = K + head_base;
    const half_t* vp = V + head_base;
    const int q0 = nblk * 64 + w * 16;          // this wave's query rows

    // Q fragments (16 x 64), loaded once from global, A-layout
    v16h aq[2];
    #pragma unroll
    for (int c = 0; c < 2; ++c)
        aq[c] = frag_ld(frag_ptr(qp, q0, DHEAD, c * 32, lane));

    v8f  oacc[4] = {};
    float mrun[8], lrun[8];
    #pragma unroll
    for (int r = 0; r < 8; ++r) { mrun[r] = -INFINITY; lrun[r] = 0.0f; }

    half_t* pw = lds_p + w * 16 * KSTR;

    for (int m0 = 0; m0 < SEQ; m0 += 64) {
        // K tile (64x64 f16, row-major): async DMA global -> LDS
        #pragma unroll
        for (int i = 0; i < 4; ++i) {
            int idx = i * 128 + tid;
            int m = idx >> 3, seg = idx & 7;     // 8 x 16B per 128B row
            async_ld_b128((unsigned)(size_t)&lds_k[m * KSTR + seg * 8],
                          kp + (size_t)(m0 + m) * DHEAD + seg * 8);
        }
        // V tile staged transposed ([dh][m]) so PV B-frag reads are contiguous
        #pragma unroll
        for (int i = 0; i < 16; ++i) {
            int idx = i * 128 + tid;
            int m = idx >> 5, d2 = (idx & 31) * 2;
            union { unsigned int u; half_t h[2]; } vv;
            vv.u = *(const unsigned int*)&vp[(size_t)(m0 + m) * DHEAD + d2];
            lds_vt[(d2 + 0) * KSTR + m] = vv.h[0];
            lds_vt[(d2 + 1) * KSTR + m] = vv.h[1];
        }
        wait_async0();
        __syncthreads();

        // S = Q @ K^T  (K rows are the B-matrix N dim; dh contiguous in LDS)
        v8f sacc[4];
        #pragma unroll
        for (int nt = 0; nt < 4; ++nt) {
            v8f z = {};
            z = wmma_f16(aq[0], frag_ld(frag_ptr(lds_k, nt * 16, KSTR, 0,  lane)), z);
            sacc[nt] = wmma_f16(aq[1], frag_ld(frag_ptr(lds_k, nt * 16, KSTR, 32, lane)), z);
        }

        // online softmax per query row r (+ hs*8)
        float alpha[8];
        #pragma unroll
        for (int r = 0; r < 8; ++r) {
            const int qrow = q0 + r + hs * 8;
            const unsigned char* mrow =
                mask + (size_t)bb * SEQ * SEQ + (size_t)qrow * SEQ + m0;
            float sv[4], mk[4];
            float rmax = -INFINITY;
            #pragma unroll
            for (int nt = 0; nt < 4; ++nt) {
                float s = ((const float*)&sacc[nt])[r] * SCALE;
                mk[nt] = mrow[nt * 16 + colid] ? 1.0f : 0.0f;
                if (mk[nt] == 0.0f) s = -1e30f;
                sv[nt] = s;
                rmax = fmaxf(rmax, s);
            }
            rmax = hmax16(rmax);
            const float mnew = fmaxf(mrun[r], rmax);
            alpha[r] = __expf(mrun[r] - mnew);
            mrun[r] = mnew;
            float rs = 0.0f;
            #pragma unroll
            for (int nt = 0; nt < 4; ++nt) {
                const float p = mk[nt] * __expf(sv[nt] - mnew);
                rs += p;
                pw[(r + hs * 8) * KSTR + nt * 16 + colid] = (half_t)p;
            }
            lrun[r] = lrun[r] * alpha[r] + hsum16(rs);
        }

        // rescale running output, then O += P @ V
        #pragma unroll
        for (int nt = 0; nt < 4; ++nt) {
            float* of = (float*)&oacc[nt];
            #pragma unroll
            for (int r = 0; r < 8; ++r) of[r] *= alpha[r];
        }
        v16h ap[2];
        #pragma unroll
        for (int c = 0; c < 2; ++c)
            ap[c] = frag_ld(frag_ptr(pw, 0, KSTR, c * 32, lane));
        #pragma unroll
        for (int nt = 0; nt < 4; ++nt) {
            oacc[nt] = wmma_f16(ap[0], frag_ld(frag_ptr(lds_vt, nt * 16, KSTR, 0,  lane)), oacc[nt]);
            oacc[nt] = wmma_f16(ap[1], frag_ld(frag_ptr(lds_vt, nt * 16, KSTR, 32, lane)), oacc[nt]);
        }
        __syncthreads();
    }

    // finalize: divide by row sum, store to (B, N, D) f16 for final projection
    #pragma unroll
    for (int r = 0; r < 8; ++r) {
        const float rl = lrun[r] > 0.0f ? 1.0f / lrun[r] : 0.0f;
        const int qrow = q0 + r + hs * 8;
        #pragma unroll
        for (int nt = 0; nt < 4; ++nt) {
            const float val = ((const float*)&oacc[nt])[r] * rl;
            const int d = hh * DHEAD + nt * 16 + colid;
            Wout[((size_t)bb * SEQ + qrow) * DMODEL + d] = (half_t)val;
        }
    }
}

// ---------------------------------------------------------------------------
extern "C" void kernel_launch(void* const* d_in, const int* in_sizes, int n_in,
                              void* d_out, int out_size, void* d_ws, size_t ws_size,
                              hipStream_t stream) {
    const float* x_kv = (const float*)d_in[0];
    const float* x_q  = (const float*)d_in[1];
    const unsigned char* msk = (const unsigned char*)d_in[2];
    const float* Wq = (const float*)d_in[3];
    const float* bq = (const float*)d_in[4];
    const float* Wk = (const float*)d_in[5];
    const float* bk = (const float*)d_in[6];
    const float* Wv = (const float*)d_in[7];
    const float* bv = (const float*)d_in[8];
    const float* Wp = (const float*)d_in[9];
    const float* bp = (const float*)d_in[10];
    float* out = (float*)d_out;

    const size_t per = (size_t)BATCH * NHEAD * SEQ * DHEAD;  // 4,194,304 halves
    half_t* qw = (half_t*)d_ws;
    half_t* kw = qw + per;
    half_t* vw = kw + per;
    half_t* ww = vw + per;   // weighted (B, N, D)

    dim3 ggrid(DMODEL / 128, (BATCH * SEQ) / 128);  // 8 x 32
    dim3 gblk(256);

    gemm_bias_wmma<float, half_t, true><<<ggrid, gblk, 0, stream>>>(x_q,  Wq, bq, qw);
    gemm_bias_wmma<float, half_t, true><<<ggrid, gblk, 0, stream>>>(x_kv, Wk, bk, kw);
    gemm_bias_wmma<float, half_t, true><<<ggrid, gblk, 0, stream>>>(x_kv, Wv, bv, vw);

    dim3 agrid(SEQ / 64, NHEAD, BATCH);             // 32 x 16 x 2
    attn_wmma<<<agrid, dim3(128), 0, stream>>>(qw, kw, vw, msk, ww);

    gemm_bias_wmma<half_t, float, false><<<ggrid, gblk, 0, stream>>>(ww, Wp, bp, out);
}
